// StructureLayer_23596550324668
// MI455X (gfx1250) — matrix operand order repaired and checked by reading, not compiled, not run on hardware
//
#include <hip/hip_runtime.h>
#include <hip/hip_bf16.h>
#include <math.h>

typedef __attribute__((ext_vector_type(16))) _Float16 v16h;
typedef __attribute__((ext_vector_type(8)))  float    v8f;

#define NRES 768
#define CS   384
#define N2   (768*768)

// ---------------------------------------------------------------------------
// Generic batched strided GEMM via V_WMMA_F32_16X16X32_F16 (wave32).
// C[b] = alpha * A[b] @ (B[b] or B[b]^T) + bias[n] + resid[b]   (opt. relu)
// Out-of-range M rows / N cols are handled by ADDRESS CLAMPING only (their
// products land in unstored D rows/cols); only the K tail needs zero fill.
// One wave computes one 16x16 C tile; block = 4 waves covering 64 cols.
// ---------------------------------------------------------------------------
struct GemmP {
    const float* A; const float* B; const float* bias; const float* resid; float* C;
    int M, N, K;
    int lda, ldb, ldc;
    long long sA, sB, sC, sRes;
    int relu, has_bias, has_res;
    float alpha;
};

template<bool TRANSB>
__global__ void __launch_bounds__(128) gemm_wmma_k(GemmP g) {
    const int wave = threadIdx.x >> 5;
    const int lane = threadIdx.x & 31;
    const int m0 = blockIdx.x * 16;
    const int n0 = blockIdx.y * 64 + wave * 16;
    if (m0 >= g.M || n0 >= g.N) return;               // wave-uniform exit
    const long long b = blockIdx.z;
    const float* Ap = g.A + b * g.sA;
    const float* Bp = g.B + b * g.sB;
    float*       Cp = g.C + b * g.sC;

    const int row = lane & 15;
    const int kA  = (lane >> 4) * 16;  // A halves 0-15 -> K=kA+0..15
    const int kB  = (lane >> 4) * 16;  // B halves 0-15 -> K=kB+0..15

    const int mmc = (m0 + row < g.M) ? (m0 + row) : (g.M - 1);  // clamped A row
    const int nnc = (n0 + row < g.N) ? (n0 + row) : (g.N - 1);  // clamped B col
    const float* Arow = Ap + (long long)mmc * g.lda;

    v8f acc = {0.f,0.f,0.f,0.f,0.f,0.f,0.f,0.f};

    int k0 = 0;
    for (; k0 + 32 <= g.K; k0 += 32) {
        // ---- A tile: one contiguous 16-float run -> 4x b128 loads ----
        const float4 a0 = *(const float4*)(Arow + k0 + kA);
        const float4 a1 = *(const float4*)(Arow + k0 + kA + 4);
        const float4 a2 = *(const float4*)(Arow + k0 + kA + 8);
        const float4 a3 = *(const float4*)(Arow + k0 + kA + 12);
        v16h av;
        av[0]=(_Float16)a0.x; av[1]=(_Float16)a0.y; av[2]=(_Float16)a0.z; av[3]=(_Float16)a0.w;
        av[4]=(_Float16)a1.x; av[5]=(_Float16)a1.y; av[6]=(_Float16)a1.z; av[7]=(_Float16)a1.w;
        av[8]=(_Float16)a2.x; av[9]=(_Float16)a2.y; av[10]=(_Float16)a2.z; av[11]=(_Float16)a2.w;
        av[12]=(_Float16)a3.x; av[13]=(_Float16)a3.y; av[14]=(_Float16)a3.z; av[15]=(_Float16)a3.w;

        v16h bv;
        if (TRANSB) {
            const float* Brow = Bp + (long long)nnc * g.ldb + k0 + kB;
            const float4 b0 = *(const float4*)(Brow + 0);
            const float4 b1 = *(const float4*)(Brow + 4);
            const float4 b2 = *(const float4*)(Brow + 8);
            const float4 b3 = *(const float4*)(Brow + 12);
            bv[0]=(_Float16)b0.x; bv[1]=(_Float16)b0.y; bv[2]=(_Float16)b0.z; bv[3]=(_Float16)b0.w;
            bv[4]=(_Float16)b1.x; bv[5]=(_Float16)b1.y; bv[6]=(_Float16)b1.z; bv[7]=(_Float16)b1.w;
            bv[8]=(_Float16)b2.x; bv[9]=(_Float16)b2.y; bv[10]=(_Float16)b2.z; bv[11]=(_Float16)b2.w;
            bv[12]=(_Float16)b3.x; bv[13]=(_Float16)b3.y; bv[14]=(_Float16)b3.z; bv[15]=(_Float16)b3.w;
        } else {
            const float* Bcol = Bp + nnc;
            #pragma unroll
            for (int hj = 0; hj < 16; ++hj)
                bv[hj] = (_Float16)Bcol[(long long)(k0 + kB + hj) * g.ldb];
        }
        acc = __builtin_amdgcn_wmma_f32_16x16x32_f16(false, av, false, bv,
                                                     (short)0, acc, false, false);
    }

    if (k0 < g.K) {  // K tail (< 32 remaining): branchless clamp + select-zero
        v16h av, bv;
        #pragma unroll
        for (int hj = 0; hj < 16; ++hj) {
            const int kk  = k0 + kA + hj;
            const int kkc = (kk < g.K) ? kk : (g.K - 1);
            float x = Arow[kkc];
            if (kk >= g.K) x = 0.f;
            av[hj] = (_Float16)x;
        }
        #pragma unroll
        for (int hj = 0; hj < 16; ++hj) {
            const int kk  = k0 + kB + hj;
            const int kkc = (kk < g.K) ? kk : (g.K - 1);
            float x = TRANSB ? Bp[(long long)nnc * g.ldb + kkc]
                             : Bp[(long long)kkc * g.ldb + nnc];
            if (kk >= g.K) x = 0.f;
            bv[hj] = (_Float16)x;
        }
        acc = __builtin_amdgcn_wmma_f32_16x16x32_f16(false, av, false, bv,
                                                     (short)0, acc, false, false);
    }

    const int nn = n0 + (lane & 15);
    if (nn >= g.N) return;
    const float bval = g.has_bias ? g.bias[nn] : 0.f;
    #pragma unroll
    for (int r = 0; r < 8; ++r) {
        const int mm = m0 + r + (lane >> 4) * 8;
        if (mm < g.M) {
            float v = acc[r] * g.alpha + bval;
            if (g.has_res) v += g.resid[b * g.sRes + (long long)mm * g.ldc + nn];
            if (g.relu && v < 0.f) v = 0.f;
            Cp[(long long)mm * g.ldc + nn] = v;
        }
    }
}

// ---------------------------------------------------------------------------
// Rotate local points to global frame: out = R * x + t.
// qp/kp: 48 points per residue, vp: 96. block = 192 threads, grid = N.
// ---------------------------------------------------------------------------
__global__ void __launch_bounds__(192) rotate_pts_k(
    const float* qp, const float* kp, const float* vp,
    const float* rots, const float* trans,
    float* qpg, float* kpg, float* vpg) {
    const int i = blockIdx.x, t = threadIdx.x;
    const float* R = rots + i * 9;
    const float* T = trans + i * 3;
    const float* src; float* dst; int idx;
    if (t < 48)      { src = qp + i * 144; dst = qpg + i * 144; idx = t; }
    else if (t < 96) { src = kp + i * 144; dst = kpg + i * 144; idx = t - 48; }
    else             { src = vp + i * 288; dst = vpg + i * 288; idx = t - 96; }
    const float x = src[idx * 3 + 0], y = src[idx * 3 + 1], z = src[idx * 3 + 2];
    #pragma unroll
    for (int r = 0; r < 3; ++r)
        dst[idx * 3 + r] = R[r * 3 + 0] * x + R[r * 3 + 1] * y + R[r * 3 + 2] * z + T[r];
}

// ---------------------------------------------------------------------------
// Block reductions (256 threads, 8 waves of 32)
// ---------------------------------------------------------------------------
__device__ inline float block_reduce_max256(float v, float* sm) {
    #pragma unroll
    for (int o = 16; o > 0; o >>= 1) v = fmaxf(v, __shfl_xor(v, o, 32));
    if ((threadIdx.x & 31) == 0) sm[threadIdx.x >> 5] = v;
    __syncthreads();
    if (threadIdx.x < 8) {
        float x = sm[threadIdx.x];
        #pragma unroll
        for (int o = 4; o > 0; o >>= 1) x = fmaxf(x, __shfl_xor(x, o, 32));
        if (threadIdx.x == 0) sm[8] = x;
    }
    __syncthreads();
    float r = sm[8];
    __syncthreads();
    return r;
}

__device__ inline float block_reduce_sum256(float v, float* sm) {
    #pragma unroll
    for (int o = 16; o > 0; o >>= 1) v += __shfl_xor(v, o, 32);
    if ((threadIdx.x & 31) == 0) sm[threadIdx.x >> 5] = v;
    __syncthreads();
    if (threadIdx.x < 8) {
        float x = sm[threadIdx.x];
        #pragma unroll
        for (int o = 4; o > 0; o >>= 1) x += __shfl_xor(x, o, 32);
        if (threadIdx.x == 0) sm[8] = x;
    }
    __syncthreads();
    float r = sm[8];
    __syncthreads();
    return r;
}

// ---------------------------------------------------------------------------
// Softmax over j of: qk_logits + sqrt(1/3)*bias - 0.5*gamma_h*d2 + mask_term.
// In place on a[h][i][j]. grid = (N, H), block = 256 (3 j per thread).
// ---------------------------------------------------------------------------
__global__ void __launch_bounds__(256) attn_softmax_k(
    float* a, const float* bias_ws, const float* qpg, const float* kpg,
    const float* head_w, const float* mask) {
    __shared__ float sred[9];
    const int i = blockIdx.x, h = blockIdx.y, t = threadIdx.x;
    float qp[12];
    #pragma unroll
    for (int u = 0; u < 12; ++u) qp[u] = qpg[i * 144 + h * 12 + u];
    const float gamma = log1pf(expf(head_w[h])) * 0.13608276348795434f; // sqrt(1/54)
    const float mi = mask[i];
    float lv[3];
    float mx = -1e30f;
    #pragma unroll
    for (int r = 0; r < 3; ++r) {
        const int j = t + r * 256;
        float l = a[(long long)h * N2 + (long long)i * 768 + j];
        l += 0.5773502691896258f * bias_ws[((long long)i * 768 + j) * 12 + h];
        float d2 = 0.f;
        #pragma unroll
        for (int p = 0; p < 4; ++p) {
            float dx = qp[p*3+0] - kpg[j * 144 + h * 12 + p*3+0];
            float dy = qp[p*3+1] - kpg[j * 144 + h * 12 + p*3+1];
            float dz = qp[p*3+2] - kpg[j * 144 + h * 12 + p*3+2];
            d2 += dx*dx + dy*dy + dz*dz;
        }
        l -= 0.5f * gamma * d2;
        l += (mi * mask[j] - 1.f) * 1e9f;
        lv[r] = l;
        mx = fmaxf(mx, l);
    }
    mx = block_reduce_max256(mx, sred);
    float sum = 0.f;
    #pragma unroll
    for (int r = 0; r < 3; ++r) { lv[r] = expf(lv[r] - mx); sum += lv[r]; }
    sum = block_reduce_sum256(sum, sred);
    const float inv = 1.f / sum;
    #pragma unroll
    for (int r = 0; r < 3; ++r)
        a[(long long)h * N2 + (long long)i * 768 + t + r * 256] = lv[r] * inv;
}

// ---------------------------------------------------------------------------
// Assemble feat = [o(192) | o_pt(288) | o_pt_norm(96) | o_pair(1536)] = 2112
// o_pt = R^T (o_pt_g - t).  grid = N, block = 256.
// ---------------------------------------------------------------------------
__global__ void __launch_bounds__(256) assemble_feat_k(
    const float* o, const float* optg, const float* opair,
    const float* rots, const float* trans, float* feat) {
    const int i = blockIdx.x, t = threadIdx.x;
    float* f = feat + (long long)i * 2112;
    for (int c = t; c < 192; c += 256) f[c] = o[i * 192 + c];
    if (t < 96) {
        const float* R = rots + i * 9;
        const float gx = optg[i * 288 + t * 3 + 0] - trans[i * 3 + 0];
        const float gy = optg[i * 288 + t * 3 + 1] - trans[i * 3 + 1];
        const float gz = optg[i * 288 + t * 3 + 2] - trans[i * 3 + 2];
        const float lx = R[0]*gx + R[3]*gy + R[6]*gz;   // R^T * g
        const float ly = R[1]*gx + R[4]*gy + R[7]*gz;
        const float lz = R[2]*gx + R[5]*gy + R[8]*gz;
        f[192 + t*3 + 0] = lx; f[192 + t*3 + 1] = ly; f[192 + t*3 + 2] = lz;
        f[480 + t] = sqrtf(lx*lx + ly*ly + lz*lz + 1e-8f);
    }
    for (int c = t; c < 1536; c += 256) f[576 + c] = opair[(long long)i * 1536 + c];
}

// ---------------------------------------------------------------------------
// Row layer-norm over CS=384. grid = N rows, block = 128.
// ---------------------------------------------------------------------------
__global__ void __launch_bounds__(128) layernorm_k(
    const float* x, const float* g, const float* b, float* y) {
    __shared__ float s1[4], s2c[4], sm[2];
    const int row = blockIdx.x, t = threadIdx.x;
    const float* xr = x + (long long)row * CS;
    float s = 0.f, q = 0.f;
    for (int c = t; c < CS; c += 128) { float v = xr[c]; s += v; q += v * v; }
    #pragma unroll
    for (int o = 16; o > 0; o >>= 1) { s += __shfl_xor(s, o, 32); q += __shfl_xor(q, o, 32); }
    if ((t & 31) == 0) { s1[t >> 5] = s; s2c[t >> 5] = q; }
    __syncthreads();
    if (t == 0) {
        float a = 0.f, bb = 0.f;
        for (int u = 0; u < 4; ++u) { a += s1[u]; bb += s2c[u]; }
        const float m = a / CS;
        sm[0] = m;
        sm[1] = rsqrtf(bb / CS - m * m + 1e-5f);
    }
    __syncthreads();
    const float m = sm[0], inv = sm[1];
    for (int c = t; c < CS; c += 128)
        y[(long long)row * CS + c] = (xr[c] - m) * inv * g[c] + b[c];
}

// ---------------------------------------------------------------------------
// Final frame update: quat normalize, r_upd, new_rots = R @ r_upd,
// new_trans = R @ u[3:] + t.  One thread per residue.
// ---------------------------------------------------------------------------
__global__ void __launch_bounds__(256) final_k(
    const float* upd, const float* rots, const float* trans,
    float* out_rots, float* out_trans) {
    const int i = blockIdx.x * blockDim.x + threadIdx.x;
    if (i >= NRES) return;
    const float* u = upd + i * 6;
    float qa = 1.f, qb = u[0], qc = u[1], qd = u[2];
    const float inv = rsqrtf(qa*qa + qb*qb + qc*qc + qd*qd);
    qa *= inv; qb *= inv; qc *= inv; qd *= inv;
    const float Ru[9] = {
        qa*qa + qb*qb - qc*qc - qd*qd, 2.f*(qb*qc - qa*qd),           2.f*(qb*qd + qa*qc),
        2.f*(qb*qc + qa*qd),           qa*qa - qb*qb + qc*qc - qd*qd, 2.f*(qc*qd - qa*qb),
        2.f*(qb*qd - qa*qc),           2.f*(qc*qd + qa*qb),           qa*qa - qb*qb - qc*qc + qd*qd };
    const float* R = rots + i * 9;
    #pragma unroll
    for (int r = 0; r < 3; ++r)
        #pragma unroll
        for (int c = 0; c < 3; ++c) {
            float acc = 0.f;
            #pragma unroll
            for (int k = 0; k < 3; ++k) acc += R[r*3+k] * Ru[k*3+c];
            out_rots[i * 9 + r * 3 + c] = acc;
        }
    #pragma unroll
    for (int r = 0; r < 3; ++r) {
        float acc = 0.f;
        #pragma unroll
        for (int c = 0; c < 3; ++c) acc += R[r*3+c] * u[3 + c];
        out_trans[i * 3 + r] = acc + trans[i * 3 + r];
    }
}

// ---------------------------------------------------------------------------
// Host launcher
// ---------------------------------------------------------------------------
extern "C" void kernel_launch(void* const* d_in, const int* in_sizes, int n_in,
                              void* d_out, int out_size, void* d_ws, size_t ws_size,
                              hipStream_t stream) {
    (void)in_sizes; (void)n_in; (void)out_size; (void)ws_size;

    const float* s     = (const float*)d_in[0];
    const float* p     = (const float*)d_in[1];
    const float* rots  = (const float*)d_in[2];
    const float* trans = (const float*)d_in[3];
    const float* mask  = (const float*)d_in[4];
    const float* wq  = (const float*)d_in[5];  const float* bq  = (const float*)d_in[6];
    const float* wk  = (const float*)d_in[7];  const float* bk  = (const float*)d_in[8];
    const float* wv  = (const float*)d_in[9];  const float* bv  = (const float*)d_in[10];
    const float* wqp = (const float*)d_in[11]; const float* bqp = (const float*)d_in[12];
    const float* wkp = (const float*)d_in[13]; const float* bkp = (const float*)d_in[14];
    const float* wvp = (const float*)d_in[15]; const float* bvp = (const float*)d_in[16];
    const float* wpb = (const float*)d_in[17]; const float* bpb = (const float*)d_in[18];
    const float* head_w = (const float*)d_in[19];
    const float* wo  = (const float*)d_in[20]; const float* bo  = (const float*)d_in[21];
    const float* ln1_g = (const float*)d_in[22]; const float* ln1_b = (const float*)d_in[23];
    const float* tw1 = (const float*)d_in[24]; const float* tb1 = (const float*)d_in[25];
    const float* tw2 = (const float*)d_in[26]; const float* tb2 = (const float*)d_in[27];
    const float* tw3 = (const float*)d_in[28]; const float* tb3 = (const float*)d_in[29];
    const float* lnt_g = (const float*)d_in[30]; const float* lnt_b = (const float*)d_in[31];
    const float* wbb = (const float*)d_in[32]; const float* bbb = (const float*)d_in[33];

    float* ws = (float*)d_ws;
    size_t off = 0;
    auto take = [&](size_t n) { float* r = ws + off; off += n; return r; };

    float* Q    = take(768 * 192);
    float* Kw   = take(768 * 192);
    float* V    = take(768 * 192);
    float* QP   = take(768 * 144);
    float* KP   = take(768 * 144);
    float* VP   = take(768 * 288);
    float* QPG  = take(768 * 144);
    float* KPG  = take(768 * 144);
    float* VPG  = take(768 * 288);
    float* BIAS = take((size_t)768 * 768 * 12);   // [i][j][h]
    float* Aws  = take((size_t)12 * 768 * 768);   // logits -> softmax in place
    float* O    = take(768 * 192);
    float* OPTG = take(768 * 288);
    float* OPAIR= take(768 * 1536);
    float* FEAT = take(768 * 2112);
    float* S1   = take(768 * 384);
    float* SLN  = take(768 * 384);
    float* H1w  = take(768 * 384);
    float* H2w  = take(768 * 384);
    float* S2   = take(768 * 384);
    float* UPD  = take(768 * 6);

    auto gemm = [&](const float* A, int lda, long long sA,
                    const float* B, int ldb, long long sB, int transB,
                    float* C, int ldc, long long sC,
                    int M, int Nn, int Kk, int batch, float alpha,
                    const float* bias, const float* resid, long long sRes, int relu) {
        GemmP g;
        g.A = A; g.B = B; g.bias = bias; g.resid = resid; g.C = C;
        g.M = M; g.N = Nn; g.K = Kk;
        g.lda = lda; g.ldb = ldb; g.ldc = ldc;
        g.sA = sA; g.sB = sB; g.sC = sC; g.sRes = sRes;
        g.relu = relu;
        g.has_bias = (bias != nullptr); g.has_res = (resid != nullptr);
        g.alpha = alpha;
        dim3 grid((M + 15) / 16, (Nn + 63) / 64, batch);
        if (transB) gemm_wmma_k<true ><<<grid, dim3(128), 0, stream>>>(g);
        else        gemm_wmma_k<false><<<grid, dim3(128), 0, stream>>>(g);
    };

    // --- projections (768 x K=384) ---
    gemm(s, 384, 0, wq,  192, 0, 0, Q,  192, 0, 768, 192, 384, 1, 1.f, bq,  nullptr, 0, 0);
    gemm(s, 384, 0, wk,  192, 0, 0, Kw, 192, 0, 768, 192, 384, 1, 1.f, bk,  nullptr, 0, 0);
    gemm(s, 384, 0, wv,  192, 0, 0, V,  192, 0, 768, 192, 384, 1, 1.f, bv,  nullptr, 0, 0);
    gemm(s, 384, 0, wqp, 144, 0, 0, QP, 144, 0, 768, 144, 384, 1, 1.f, bqp, nullptr, 0, 0);
    gemm(s, 384, 0, wkp, 144, 0, 0, KP, 144, 0, 768, 144, 384, 1, 1.f, bkp, nullptr, 0, 0);
    gemm(s, 384, 0, wvp, 288, 0, 0, VP, 288, 0, 768, 288, 384, 1, 1.f, bvp, nullptr, 0, 0);

    rotate_pts_k<<<dim3(768), dim3(192), 0, stream>>>(QP, KP, VP, rots, trans, QPG, KPG, VPG);

    // --- bias = p @ wpb + bpb, batched over i (reads p once: 302 MB) ---
    gemm(p, 128, 98304, wpb, 12, 0, 0, BIAS, 12, 9216,
         768, 12, 128, 768, 1.f, bpb, nullptr, 0, 0);

    // --- logits = (q . k) * sqrt(1/48), batched over h (K=16 tail path) ---
    gemm(Q, 192, 16, Kw, 192, 16, 1, Aws, 768, (long long)N2,
         768, 768, 16, 12, 0.14433756729740643f, nullptr, nullptr, 0, 0);

    attn_softmax_k<<<dim3(768, 12), dim3(256), 0, stream>>>(Aws, BIAS, QPG, KPG, head_w, mask);

    // --- o = a @ v (per h) ---
    gemm(Aws, 768, (long long)N2, V, 192, 16, 0, O, 192, 16,
         768, 16, 768, 12, 1.f, nullptr, nullptr, 0, 0);
    // --- o_pt_g = a @ vp_g (per h) ---
    gemm(Aws, 768, (long long)N2, VPG, 288, 24, 0, OPTG, 288, 24,
         768, 24, 768, 12, 1.f, nullptr, nullptr, 0, 0);
    // --- o_pair = a @ p, batched over i (second 302 MB read of p) ---
    gemm(Aws, (int)N2, 768, p, 128, 98304, 0, OPAIR, 128, 1536,
         12, 128, 768, 768, 1.f, nullptr, nullptr, 0, 0);

    assemble_feat_k<<<dim3(768), dim3(256), 0, stream>>>(O, OPTG, OPAIR, rots, trans, FEAT);

    // --- s1 = s + feat @ wo + bo ---
    gemm(FEAT, 2112, 0, wo, 384, 0, 0, S1, 384, 0,
         768, 384, 2112, 1, 1.f, bo, s, 0, 0);
    layernorm_k<<<dim3(768), dim3(128), 0, stream>>>(S1, ln1_g, ln1_b, SLN);

    // --- transition MLP ---
    gemm(SLN, 384, 0, tw1, 384, 0, 0, H1w, 384, 0, 768, 384, 384, 1, 1.f, tb1, nullptr, 0, 1);
    gemm(H1w, 384, 0, tw2, 384, 0, 0, H2w, 384, 0, 768, 384, 384, 1, 1.f, tb2, nullptr, 0, 1);
    gemm(H2w, 384, 0, tw3, 384, 0, 0, S2,  384, 0, 768, 384, 384, 1, 1.f, tb3, SLN,     0, 0);

    float* sout = (float*)d_out;                       // output region 0: s (768*384)
    layernorm_k<<<dim3(768), dim3(128), 0, stream>>>(S2, lnt_g, lnt_b, sout);

    // --- upd = s_out @ wbb + bbb ---
    gemm(sout, 384, 0, wbb, 6, 0, 0, UPD, 6, 0, 768, 6, 384, 1, 1.f, bbb, nullptr, 0, 0);

    final_k<<<dim3(3), dim3(256), 0, stream>>>(UPD, rots, trans,
                                               sout + 768 * 384,            // new_rots
                                               sout + 768 * 384 + 768 * 9); // new_trans
}
